// Attention_11613591568557
// MI455X (gfx1250) — matrix operand order repaired and checked by reading, not compiled, not run on hardware
//
#include <hip/hip_runtime.h>
#include <hip/hip_bf16.h>
#include <math.h>

// ---------------- problem constants (B=16, S=4096, D=1024) ----------------
#define BATCH   16
#define SLEN    4096
#define DDIM    1024
#define MROWS   65536          // B*S
#define M_TILE  128            // rows per block in the fused GEMM
#define LDA     1032           // padded LDS row stride (elements) -> conflict-free b128 reads

typedef __bf16 v16bf __attribute__((ext_vector_type(16)));
typedef float  v8f   __attribute__((ext_vector_type(8)));

__device__ __forceinline__ unsigned short f32_to_bf16(float f) {
    unsigned int u = __float_as_uint(f);
    u += 0x7FFFu + ((u >> 16) & 1u);      // round-to-nearest-even
    return (unsigned short)(u >> 16);
}

// Branch-free tanh: single V_TANH_F32 on CDNA5, short trans sequence otherwise.
__device__ __forceinline__ float fast_tanh(float x) {
#if __has_builtin(__builtin_amdgcn_tanhf)
    return __builtin_amdgcn_tanhf(x);
#elif __has_builtin(__builtin_amdgcn_exp2f) && __has_builtin(__builtin_amdgcn_rcpf)
    // tanh(x) = 1 - 2/(exp(2x)+1); exp(2x) = exp2(x * 2*log2(e))
    float e = __builtin_amdgcn_exp2f(2.8853900817779268f * x);
    float r = __builtin_amdgcn_rcpf(e + 1.0f);
    return 1.0f - 2.0f * r;
#else
    float e = __expf(2.0f * x);
    return 1.0f - 2.0f / (e + 1.0f);
#endif
}

// ---------------- K0: W_h f32 -> bf16 (2MB, stays L2-resident) ----------------
__global__ __launch_bounds__(256) void wh_bf16_kernel(const float* __restrict__ wh,
                                                      unsigned short* __restrict__ whb) {
    int idx = blockIdx.x * 256 + threadIdx.x;   // 1M elements
    whb[idx] = f32_to_bf16(wh[idx]);
}

// ---------------- K1: dec_feat[b][e] = dh[b]·W_s[e] + b_s[e] (tiny) ----------------
__global__ __launch_bounds__(256) void dec_feat_kernel(const float* __restrict__ dh,
                                                       const float* __restrict__ Ws,
                                                       const float* __restrict__ bs,
                                                       float* __restrict__ decFeat) {
    int gw   = (blockIdx.x * 256 + threadIdx.x) >> 5;   // one wave per (b,e)
    int lane = threadIdx.x & 31;
    int e = gw & (DDIM - 1);
    int b = gw >> 10;
    const float* wr = Ws + (size_t)e * DDIM;
    const float* hr = dh + (size_t)b * DDIM;
    float acc = 0.f;
    for (int d = lane; d < DDIM; d += 32) acc += wr[d] * hr[d];
    #pragma unroll
    for (int off = 16; off; off >>= 1) acc += __shfl_xor(acc, off, 32);
    if (lane == 0) decFeat[b * DDIM + e] = acc + bs[e];
}

// ---------------- K2: fused enc@W_h^T + tanh epilogue -> scores ----------------
// Each block: 128 rows of (b,s); 8 waves, one 16-row M-subtile per wave.
// A tile staged once (f32->bf16). W_h panel double-buffered through registers:
// next 16x1024 bf16 panel is fetched from (L2-resident) global while the
// current panel's 32 WMMAs execute; tile boundary just dumps regs to LDS.
__global__ __launch_bounds__(256, 1) void attn_scores_kernel(
        const float* __restrict__ enc, const float* __restrict__ cov,
        const unsigned short* __restrict__ whb, const float* __restrict__ decFeat,
        const float* __restrict__ wc, const float* __restrict__ vvec,
        float* __restrict__ scores) {
    __shared__ unsigned short Alds[M_TILE * LDA];   // 264192 B
    __shared__ unsigned short Blds[16 * LDA];       //  33024 B
    __shared__ float decL[DDIM], wcL[DDIM], vL[DDIM];  // 12288 B  (total ~309.5 KB)

    const int tid  = threadIdx.x;
    const int g0   = blockIdx.x * M_TILE;          // first global row (b,s)
    const int b    = g0 >> 12;                     // S=4096 rows per batch
    const int s0   = g0 & (SLEN - 1);

    // epilogue vectors -> LDS
    for (int i = tid; i < DDIM; i += 256) {
        decL[i] = decFeat[b * DDIM + i];
        wcL[i]  = wc[i];
        vL[i]   = vvec[i];
    }
    // stage A tile: 128x1024 f32 -> bf16, float4 granularity (32768 quads)
    for (int p = tid; p < (M_TILE * DDIM / 4); p += 256) {
        int row = p >> 8;                 // 256 quads per row
        int c4  = (p & 255) << 2;
        const float4 f = *reinterpret_cast<const float4*>(
                              enc + (size_t)(g0 + row) * DDIM + c4);
        uint2 packed;
        packed.x = (unsigned int)f32_to_bf16(f.x) | ((unsigned int)f32_to_bf16(f.y) << 16);
        packed.y = (unsigned int)f32_to_bf16(f.z) | ((unsigned int)f32_to_bf16(f.w) << 16);
        *reinterpret_cast<uint2*>(&Alds[row * LDA + c4]) = packed;
    }

    const int wave = tid >> 5;
    const int lane = tid & 31;
    const int half = lane >> 4;        // 0: lanes 0-15, 1: lanes 16-31
    const int mn   = lane & 15;

    // per-C-VGPR coverage value (row = wave*16 + r + 8*half)
    float covR[8];
    #pragma unroll
    for (int r = 0; r < 8; ++r)
        covR[r] = cov[b * SLEN + s0 + wave * 16 + r + half * 8];

    float scoreAcc[8] = {0.f,0.f,0.f,0.f,0.f,0.f,0.f,0.f};

    const int aBase = (wave * 16 + mn) * LDA + half * 8;   // ISA 16-bit A layout
    const int bBase = mn * LDA + half * 8;                 // ISA 16-bit B layout

    // B-panel register buffer: 8 x uint4 per thread = one 16x1024 bf16 panel/block
    uint4 breg[8];
    #pragma unroll
    for (int j = 0; j < 8; ++j) {
        int i = tid + j * 256, row = i >> 7, q = i & 127;
        breg[j] = *(reinterpret_cast<const uint4*>(whb + (size_t)row * DDIM) + q);
    }
    __syncthreads();   // A tile + epilogue vectors ready

    for (int nt = 0; nt < DDIM / 16; ++nt) {
        // dump staged panel to LDS
        #pragma unroll
        for (int j = 0; j < 8; ++j) {
            int i = tid + j * 256, row = i >> 7, q = i & 127;
            *reinterpret_cast<uint4*>(&Blds[row * LDA + q * 8]) = breg[j];
        }
        __syncthreads();

        // prefetch next panel (overlaps with WMMA loop below)
        if (nt + 1 < DDIM / 16) {
            const unsigned short* pnext = whb + (size_t)(nt + 1) * 16 * DDIM;
            #pragma unroll
            for (int j = 0; j < 8; ++j) {
                int i = tid + j * 256, row = i >> 7, q = i & 127;
                breg[j] = *(reinterpret_cast<const uint4*>(pnext + (size_t)row * DDIM) + q);
            }
        }

        union { v8f v; float f[8]; } acc;
        acc.v = (v8f){0.f,0.f,0.f,0.f,0.f,0.f,0.f,0.f};
        #pragma unroll 4
        for (int k0 = 0; k0 < DDIM; k0 += 32) {
            union { v16bf v; uint4 q[2]; } af, bf;
            af.q[0] = *reinterpret_cast<const uint4*>(&Alds[aBase + k0]);
            af.q[1] = *reinterpret_cast<const uint4*>(&Alds[aBase + k0 + 16]);
            bf.q[0] = *reinterpret_cast<const uint4*>(&Blds[bBase + k0]);
            bf.q[1] = *reinterpret_cast<const uint4*>(&Blds[bBase + k0 + 16]);
            acc.v = __builtin_amdgcn_wmma_f32_16x16x32_bf16(
                        false, af.v, false, bf.v, (short)0, acc.v, false, false);
        }
        // fused epilogue: tanh(C + dec_feat + cov*w_c) * v, accumulate per lane
        const int col = nt * 16 + mn;
        const float dv = decL[col], wv = wcL[col], vv = vL[col];
        #pragma unroll
        for (int r = 0; r < 8; ++r)
            scoreAcc[r] += fast_tanh(acc.f[r] + dv + covR[r] * wv) * vv;
        __syncthreads();   // compute done before Blds overwritten
    }

    // cross-lane reduce within each 16-lane half (columns of one row)
    #pragma unroll
    for (int r = 0; r < 8; ++r) {
        float sv = scoreAcc[r];
        #pragma unroll
        for (int off = 1; off <= 8; off <<= 1) sv += __shfl_xor(sv, off, 32);
        if (mn == 0) scores[g0 + wave * 16 + half * 8 + r] = sv;
    }
}

// ---------------- K3: masked softmax + renorm, coverage_new, zero context ----------------
// d_out layout: context[16*1024] | attn[16*4096] | coverage_new[16*4096]
#define OUT_ATTN  (BATCH * DDIM)
#define OUT_COV   (BATCH * DDIM + BATCH * SLEN)
__global__ __launch_bounds__(256) void softmax_kernel(const float* __restrict__ scores,
                                                      const float* __restrict__ mask,
                                                      const float* __restrict__ cov,
                                                      float* __restrict__ out) {
    const int b = blockIdx.x, tid = threadIdx.x;
    __shared__ float red[256];
    float m = -INFINITY;
    for (int s = tid; s < SLEN; s += 256) m = fmaxf(m, scores[b * SLEN + s]);
    red[tid] = m; __syncthreads();
    for (int st = 128; st; st >>= 1) { if (tid < st) red[tid] = fmaxf(red[tid], red[tid + st]); __syncthreads(); }
    m = red[0]; __syncthreads();
    float sum = 0.f;
    for (int s = tid; s < SLEN; s += 256)
        sum += mask[b * SLEN + s] * expf(scores[b * SLEN + s] - m);
    red[tid] = sum; __syncthreads();
    for (int st = 128; st; st >>= 1) { if (tid < st) red[tid] += red[tid + st]; __syncthreads(); }
    const float inv = 1.f / red[0];
    for (int s = tid; s < SLEN; s += 256) {
        float a = mask[b * SLEN + s] * expf(scores[b * SLEN + s] - m) * inv;
        out[OUT_ATTN + b * SLEN + s] = a;
        out[OUT_COV  + b * SLEN + s] = cov[b * SLEN + s] + a;
    }
    for (int d = tid; d < DDIM; d += 256) out[b * DDIM + d] = 0.f;  // init context
}

// ---------------- K4: context[b,d] = sum_s attn[b,s]*enc[b,s,d] (atomic partials) ----------------
__global__ __launch_bounds__(256) void context_kernel(const float* __restrict__ enc,
                                                      float* __restrict__ out) {
    // 512 blocks = b(16) x dchunk(4) x schunk(8)
    const int sc = blockIdx.x & 7;
    const int dc = (blockIdx.x >> 3) & 3;
    const int b  = blockIdx.x >> 5;
    const int d  = dc * 256 + threadIdx.x;
    __shared__ float aa[512];
    const float* attn = out + OUT_ATTN + b * SLEN + sc * 512;
    for (int i = threadIdx.x; i < 512; i += 256) aa[i] = attn[i];
    __syncthreads();
    float acc = 0.f;
    const float* ep = enc + ((size_t)b * SLEN + sc * 512) * DDIM + d;
    for (int s = 0; s < 512; ++s) acc += aa[s] * ep[(size_t)s * DDIM];
    atomicAdd(&out[b * DDIM + d], acc);
}

// ---------------- launcher ----------------
extern "C" void kernel_launch(void* const* d_in, const int* in_sizes, int n_in,
                              void* d_out, int out_size, void* d_ws, size_t ws_size,
                              hipStream_t stream) {
    const float* enc  = (const float*)d_in[0];
    const float* dh   = (const float*)d_in[1];
    const float* mask = (const float*)d_in[2];
    const float* cov  = (const float*)d_in[3];
    const float* Wh   = (const float*)d_in[4];
    const float* Ws   = (const float*)d_in[5];
    const float* bs   = (const float*)d_in[6];
    const float* wc   = (const float*)d_in[7];
    const float* vv   = (const float*)d_in[8];
    float* out = (float*)d_out;

    unsigned short* whb = (unsigned short*)d_ws;                              // 2 MB
    float* decFeat = (float*)((char*)d_ws + (2u << 20));                      // 64 KB
    float* scores  = (float*)((char*)d_ws + (2u << 20) + (64u << 10));        // 256 KB

    wh_bf16_kernel   <<<4096, 256, 0, stream>>>(Wh, whb);
    dec_feat_kernel  <<<2048, 256, 0, stream>>>(dh, Ws, bs, decFeat);
    attn_scores_kernel<<<MROWS / M_TILE, 256, 0, stream>>>(enc, cov, whb, decFeat, wc, vv, scores);
    softmax_kernel   <<<BATCH, 256, 0, stream>>>(scores, mask, cov, out);
    context_kernel   <<<512, 256, 0, stream>>>(enc, out);
}
